// BitNetAttention_3788161155375
// MI455X (gfx1250) — matrix operand order repaired and checked by reading, not compiled, not run on hardware
//
#include <hip/hip_runtime.h>
#include <hip/hip_bf16.h>

// ---------------------------------------------------------------------------
// BitNet attention block for MI455X (gfx1250, wave32, WMMA f16 16x16x32).
// Pipeline: quantize x -> f16 QKV GEMM (WMMA + async-LDS double buffering) ->
// RoPE/split -> flash attention (WMMA QK^T and PV) -> RMSNorm+quantize ->
// f16 output GEMM (WMMA + async-LDS).
// ---------------------------------------------------------------------------

typedef _Float16 half_t;
typedef __attribute__((ext_vector_type(16))) _Float16 v16h;
typedef __attribute__((ext_vector_type(8)))  _Float16 v8h;
typedef __attribute__((ext_vector_type(8)))  float    v8f;

union AFrag { v16h v; v8h h[2]; };

// dims
#define SEQ   2048
#define HID   2560
#define NH    20
#define NKV   5
#define HD    128
#define QKVW  3840          // (NH + 2*NKV) * HD
#define KOFF  2560
#define VOFF  3200

#define NEG_INF (-3.402823466e+38f)

// LDS row pitch in halves (32 data + 8 pad; keeps 16B alignment of frag reads)
#define LPITCH 40

__device__ __forceinline__ void async_load_16B(uint32_t lds_off, const half_t* gaddr)
{
    asm volatile("global_load_async_to_lds_b128 %0, %1, off"
                 :: "v"(lds_off), "v"(gaddr) : "memory");
}

// ---------------------------------------------------------------------------
// 1) per-row absmax quantization of x -> f16 integers + per-row scale (1/s)
// ---------------------------------------------------------------------------
__global__ void __launch_bounds__(256)
quant_x_kernel(const float* __restrict__ x, half_t* __restrict__ xq,
               float* __restrict__ xscale)
{
    __shared__ float red[256];
    const int row = blockIdx.x;
    const float* xr = x + (size_t)row * HID;
    float amax = 0.f;
    for (int c = threadIdx.x; c < HID; c += 256) amax = fmaxf(amax, fabsf(xr[c]));
    red[threadIdx.x] = amax; __syncthreads();
    for (int s = 128; s > 0; s >>= 1) {
        if (threadIdx.x < s) red[threadIdx.x] = fmaxf(red[threadIdx.x], red[threadIdx.x + s]);
        __syncthreads();
    }
    const float sc = 127.0f / fmaxf(red[0], 1e-5f);
    if (threadIdx.x == 0) xscale[row] = 1.0f / sc;
    half_t* oq = xq + (size_t)row * HID;
    for (int c = threadIdx.x; c < HID; c += 256) {
        float v = fminf(fmaxf(rintf(xr[c] * sc), -128.f), 127.f);  // round-half-even like jnp.round
        oq[c] = (half_t)v;
    }
}

// ---------------------------------------------------------------------------
// 2) fp32 -> fp16 weight conversion
// ---------------------------------------------------------------------------
__global__ void __launch_bounds__(256)
cvt_f32_f16_kernel(const float* __restrict__ in, half_t* __restrict__ out, int n)
{
    int i = blockIdx.x * 256 + threadIdx.x;
    if (i < n) out[i] = (half_t)in[i];
}

// ---------------------------------------------------------------------------
// 3) WMMA GEMM: C[M,N] = rowscale[m] * (A[M,K] @ B[N,K]^T)
//    block = 256 thr = 8 waves; wave tile 16x64 -> block tile 64x128.
//    A (64x32) and B (128x32) k-slices staged in LDS via
//    GLOBAL_LOAD_ASYNC_TO_LDS_B128 (ASYNCcnt), double buffered: each thread
//    issues exactly 3 async 16B chunks per stage, so s_wait_asynccnt 0x3
//    drains the previous stage while the next is in flight.
//    Frag layouts (this code's k order): A lane (l&15)=row, K contiguous
//    from (l>>4)*16; B lane (l&15)=col, K contiguous
//    from (l>>4)*16.
// ---------------------------------------------------------------------------
__global__ void __launch_bounds__(256)
gemm_rs_kernel(const half_t* __restrict__ A, const half_t* __restrict__ B,
               const float* __restrict__ rowscale, float* __restrict__ C,
               int M, int N, int K)
{
    __shared__ __align__(16) half_t Asb[2][64  * LPITCH];
    __shared__ __align__(16) half_t Bsb[2][128 * LPITCH];

    const int tid  = threadIdx.x;
    const int lane = tid & 31;
    const int wave = tid >> 5;
    const int hi   = lane >> 4;
    const int lo16 = lane & 15;
    const int mblk = blockIdx.y * 64;
    const int nblk = blockIdx.x * 128;
    const int mloc = (wave & 3) * 16 + lo16;   // A-frag row within block tile
    const int nw   = (wave >> 2) * 64;         // wave's n-offset within block tile

    // async-copy assignments: A: chunk tid (64 rows x 4 chunks);
    // B: chunks tid and tid+256 (128 rows x 4 chunks)
    const int ar  = tid >> 2,          akc = (tid & 3) * 8;
    const int br0 = tid >> 2,          bk0 = (tid & 3) * 8;
    const int br1 = (tid + 256) >> 2,  bk1 = (tid & 3) * 8;

    const half_t* Ab = A + (size_t)(mblk + ar)  * K + akc;
    const half_t* Bb0 = B + (size_t)(nblk + br0) * K + bk0;
    const half_t* Bb1 = B + (size_t)(nblk + br1) * K + bk1;

    auto issue_stage = [&](int k0, int buf) {
        async_load_16B((uint32_t)(uintptr_t)&Asb[buf][ar  * LPITCH + akc], Ab  + k0);
        async_load_16B((uint32_t)(uintptr_t)&Bsb[buf][br0 * LPITCH + bk0], Bb0 + k0);
        async_load_16B((uint32_t)(uintptr_t)&Bsb[buf][br1 * LPITCH + bk1], Bb1 + k0);
    };

    v8f acc[4] = {};
    issue_stage(0, 0);

    int buf = 0;
    for (int k0 = 0; k0 < K; k0 += 32, buf ^= 1) {
        if (k0 + 32 < K) {
            issue_stage(k0 + 32, buf ^ 1);
            asm volatile("s_wait_asynccnt 0x3" ::: "memory");
        } else {
            asm volatile("s_wait_asynccnt 0x0" ::: "memory");
        }
        __syncthreads();

        AFrag a;
        a.h[0] = *(const v8h*)(&Asb[buf][mloc * LPITCH + hi * 16]);
        a.h[1] = *(const v8h*)(&Asb[buf][mloc * LPITCH + hi * 16 + 8]);
#pragma unroll
        for (int t = 0; t < 4; ++t) {
            v16h b = *(const v16h*)(&Bsb[buf][(nw + t * 16 + lo16) * LPITCH + hi * 16]);
            acc[t] = __builtin_amdgcn_wmma_f32_16x16x32_f16(
                false, a.v, false, b, (short)0, acc[t], false, false);
        }
        __syncthreads();   // all waves done with buf before it is overwritten
    }

#pragma unroll
    for (int t = 0; t < 4; ++t) {
#pragma unroll
        for (int i = 0; i < 8; ++i) {
            const int row = mblk + (wave & 3) * 16 + hi * 8 + i;
            const int col = nblk + nw + t * 16 + lo16;
            C[(size_t)row * N + col] = acc[t][i] * rowscale[row];
        }
    }
}

// ---------------------------------------------------------------------------
// 4) RoPE + head split.  qkv fp32 [S, 3840] ->
//      Qh f16 [NH,S,HD] (pre-scaled by 1/sqrt(HD)), Kh f16 [NKV,S,HD],
//      Vt f16 [NKV,HD,S]  (transposed so PV B-frags are contiguous loads)
// ---------------------------------------------------------------------------
__global__ void __launch_bounds__(256)
rope_split_kernel(const float* __restrict__ qkv, half_t* __restrict__ Qh,
                  half_t* __restrict__ Kh, half_t* __restrict__ Vt)
{
    const int s = blockIdx.x;
    const float* row = qkv + (size_t)s * QKVW;
    const float qsc = 0.08838834764831845f;   // 1/sqrt(128)
    for (int t = threadIdx.x; t < QKVW; t += 256) {
        const float v = row[t];
        if (t < KOFF) {                         // Q with RoPE
            const int h = t >> 7, d = t & 127, i = d & 63;
            const float invf = __powf(500000.0f, -(float)(2 * i) / 128.0f);
            float sn, cs; __sincosf((float)s * invf, &sn, &cs);
            const float rh = (d < 64) ? -row[t + 64] : row[t - 64];
            Qh[((size_t)h * SEQ + s) * HD + d] = (half_t)((v * cs + rh * sn) * qsc);
        } else if (t < VOFF) {                  // K with RoPE
            const int u = t - KOFF, h = u >> 7, d = u & 127, i = d & 63;
            const float invf = __powf(500000.0f, -(float)(2 * i) / 128.0f);
            float sn, cs; __sincosf((float)s * invf, &sn, &cs);
            const float rh = (d < 64) ? -row[t + 64] : row[t - 64];
            Kh[((size_t)h * SEQ + s) * HD + d] = (half_t)(v * cs + rh * sn);
        } else {                                // V, transposed
            const int u = t - VOFF, h = u >> 7, d = u & 127;
            Vt[((size_t)h * HD + d) * SEQ + s] = (half_t)v;
        }
    }
}

// ---------------------------------------------------------------------------
// 5) Flash attention.  grid = (SEQ/64, NH), block = 128 thr = 4 waves.
//    Wave handles 16 queries; key blocks of 32; online softmax with half-wave
//    shuffles; P converted C-frag -> A-frag via per-wave LDS tile.
// ---------------------------------------------------------------------------
__global__ void __launch_bounds__(128)
attn_kernel(const half_t* __restrict__ Qh, const half_t* __restrict__ Kh,
            const half_t* __restrict__ Vt, float* __restrict__ O)
{
    __shared__ __align__(16) half_t Pl[4][16][32];
    const int lane = threadIdx.x & 31;
    const int wave = threadIdx.x >> 5;
    const int hi = lane >> 4, lo16 = lane & 15;
    const int h   = blockIdx.y;
    const int kvh = h >> 2;                    // g = NH/NKV = 4
    const int q0  = blockIdx.x * 64 + wave * 16;

    // preload the wave's Q tile as 4 A-frags (K-dim = 128 = 4 x 32)
    AFrag qa[4];
    const half_t* qrow = Qh + ((size_t)h * SEQ + q0 + lo16) * HD + hi * 16;
#pragma unroll
    for (int kd = 0; kd < 4; ++kd) {
        qa[kd].h[0] = *(const v8h*)(qrow + kd * 32);
        qa[kd].h[1] = *(const v8h*)(qrow + kd * 32 + 8);
    }

    v8f oacc[8] = {};
    float rmax[8], rsum[8];
#pragma unroll
    for (int i = 0; i < 8; ++i) { rmax[i] = NEG_INF; rsum[i] = 0.f; }

    const half_t* Kbase = Kh + (size_t)kvh * SEQ * HD;
    const half_t* Vbase = Vt + (size_t)kvh * HD * SEQ;

    for (int kb = 0; kb < SEQ; kb += 32) {
        // S tile 16x32 = two 16x16 WMMA accumulations over head_dim
        v8f sfr[2];
#pragma unroll
        for (int kn = 0; kn < 2; ++kn) {
            v8f acc = {};
            const half_t* krow = Kbase + (size_t)(kb + kn * 16 + lo16) * HD + hi * 16;
#pragma unroll
            for (int kd = 0; kd < 4; ++kd) {
                v16h b = *(const v16h*)(krow + kd * 32);
                acc = __builtin_amdgcn_wmma_f32_16x16x32_f16(
                    false, qa[kd].v, false, b, (short)0, acc, false, false);
            }
            sfr[kn] = acc;
        }
        // online softmax; C-frag row i lives in one 16-lane half at column lo16
#pragma unroll
        for (int i = 0; i < 8; ++i) {
            float mx = fmaxf(sfr[0][i], sfr[1][i]);
            for (int off = 8; off > 0; off >>= 1) mx = fmaxf(mx, __shfl_xor(mx, off));
            const float nm   = fmaxf(rmax[i], mx);
            const float corr = __expf(rmax[i] - nm);
            const float p0 = __expf(sfr[0][i] - nm);
            const float p1 = __expf(sfr[1][i] - nm);
            float rs = p0 + p1;
            for (int off = 8; off > 0; off >>= 1) rs += __shfl_xor(rs, off);
            rsum[i] = rsum[i] * corr + rs;
            rmax[i] = nm;
#pragma unroll
            for (int t = 0; t < 8; ++t) oacc[t][i] *= corr;
            const int m = hi * 8 + i;
            Pl[wave][m][lo16]      = (half_t)p0;
            Pl[wave][m][lo16 + 16] = (half_t)p1;
        }
        // P C-frag -> A-frag through LDS (wave-private tile; backend inserts dscnt wait)
        AFrag pa;
        pa.h[0] = *(const v8h*)(&Pl[wave][lo16][hi * 16]);
        pa.h[1] = *(const v8h*)(&Pl[wave][lo16][hi * 16 + 8]);
        // O += P(16x32) @ V(32x128): 8 dim-tiles, contiguous B loads from Vt
#pragma unroll
        for (int t = 0; t < 8; ++t) {
            const half_t* vrow = Vbase + (size_t)(t * 16 + lo16) * SEQ + kb + hi * 16;
            v16h b = *(const v16h*)vrow;
            oacc[t] = __builtin_amdgcn_wmma_f32_16x16x32_f16(
                false, pa.v, false, b, (short)0, oacc[t], false, false);
        }
    }
    // normalize and write [S, NH*HD]
#pragma unroll
    for (int t = 0; t < 8; ++t) {
#pragma unroll
        for (int i = 0; i < 8; ++i) {
            const int m = hi * 8 + i;
            O[(size_t)(q0 + m) * HID + h * HD + t * 16 + lo16] = oacc[t][i] / rsum[i];
        }
    }
}

// ---------------------------------------------------------------------------
// 6) RMSNorm (in-place on attn output) + absmax re-quantize -> f16 + scale
// ---------------------------------------------------------------------------
__global__ void __launch_bounds__(256)
rmsnorm_quant_kernel(float* __restrict__ y, const float* __restrict__ nw,
                     half_t* __restrict__ yq, float* __restrict__ yscale)
{
    __shared__ float red[256];
    const int row = blockIdx.x;
    float* yr = y + (size_t)row * HID;
    float ss = 0.f;
    for (int c = threadIdx.x; c < HID; c += 256) { float v = yr[c]; ss += v * v; }
    red[threadIdx.x] = ss; __syncthreads();
    for (int s = 128; s > 0; s >>= 1) {
        if (threadIdx.x < s) red[threadIdx.x] += red[threadIdx.x + s];
        __syncthreads();
    }
    const float rms = rsqrtf(red[0] / (float)HID + 1e-5f);
    __syncthreads();
    float amax = 0.f;
    for (int c = threadIdx.x; c < HID; c += 256) {
        const float v = nw[c] * yr[c] * rms;
        yr[c] = v;
        amax = fmaxf(amax, fabsf(v));
    }
    red[threadIdx.x] = amax; __syncthreads();
    for (int s = 128; s > 0; s >>= 1) {
        if (threadIdx.x < s) red[threadIdx.x] = fmaxf(red[threadIdx.x], red[threadIdx.x + s]);
        __syncthreads();
    }
    const float sc = 127.0f / fmaxf(red[0], 1e-5f);
    if (threadIdx.x == 0) yscale[row] = 1.0f / sc;
    half_t* oq = yq + (size_t)row * HID;
    for (int c = threadIdx.x; c < HID; c += 256)
        oq[c] = (half_t)fminf(fmaxf(rintf(yr[c] * sc), -128.f), 127.f);
}

// ---------------------------------------------------------------------------
// launch
// ---------------------------------------------------------------------------
extern "C" void kernel_launch(void* const* d_in, const int* in_sizes, int n_in,
                              void* d_out, int out_size, void* d_ws, size_t ws_size,
                              hipStream_t stream)
{
    const float* x    = (const float*)d_in[0];
    const float* wqkv = (const float*)d_in[1];
    const float* wo   = (const float*)d_in[2];
    const float* nw   = (const float*)d_in[3];
    float* out = (float*)d_out;

    char* ws = (char*)d_ws;
    auto alloc = [&](size_t bytes) -> char* {
        char* p = ws; ws += (bytes + 255) & ~(size_t)255; return p;
    };
    half_t* xq      = (half_t*)alloc((size_t)SEQ * HID * sizeof(half_t));
    float*  xscale  = (float*) alloc((size_t)SEQ * sizeof(float));
    half_t* wqkv16  = (half_t*)alloc((size_t)QKVW * HID * sizeof(half_t));
    half_t* wo16    = (half_t*)alloc((size_t)HID * HID * sizeof(half_t));
    float*  qkvb    = (float*) alloc((size_t)SEQ * QKVW * sizeof(float));
    half_t* Qh      = (half_t*)alloc((size_t)NH  * SEQ * HD * sizeof(half_t));
    half_t* Kh      = (half_t*)alloc((size_t)NKV * SEQ * HD * sizeof(half_t));
    half_t* Vt      = (half_t*)alloc((size_t)NKV * HD * SEQ * sizeof(half_t));
    float*  attnout = (float*) alloc((size_t)SEQ * HID * sizeof(float));
    half_t* yq      = (half_t*)alloc((size_t)SEQ * HID * sizeof(half_t));
    float*  yscale  = (float*) alloc((size_t)SEQ * sizeof(float));

    quant_x_kernel<<<SEQ, 256, 0, stream>>>(x, xq, xscale);
    cvt_f32_f16_kernel<<<(QKVW * HID + 255) / 256, 256, 0, stream>>>(wqkv, wqkv16, QKVW * HID);
    cvt_f32_f16_kernel<<<(HID * HID + 255) / 256, 256, 0, stream>>>(wo, wo16, HID * HID);
    gemm_rs_kernel<<<dim3(QKVW / 128, SEQ / 64), 256, 0, stream>>>(
        xq, wqkv16, xscale, qkvb, SEQ, QKVW, HID);
    rope_split_kernel<<<SEQ, 256, 0, stream>>>(qkvb, Qh, Kh, Vt);
    attn_kernel<<<dim3(SEQ / 64, NH), 128, 0, stream>>>(Qh, Kh, Vt, attnout);
    rmsnorm_quant_kernel<<<SEQ, 256, 0, stream>>>(attnout, nw, yq, yscale);
    gemm_rs_kernel<<<dim3(HID / 128, SEQ / 64), 256, 0, stream>>>(
        yq, wo16, yscale, out, SEQ, HID, HID);
}
